// ALSTM_46952582480190
// MI455X (gfx1250) — matrix operand order repaired and checked by reading, not compiled, not run on hardware
//
#include <hip/hip_runtime.h>
#include <math.h>

// ---------------------------------------------------------------------------
// Adaptive-computation-time LSTM for MI455X (gfx1250).
//
// Perf model: 101 sequential GEMVs (8192x2048) dominate. ~3.4 GFLOP total ->
// purely bandwidth/latency bound. W_hh fp32 = 64MB fits in 192MB L2; we
// convert it ONCE to f16 (32MB, halving per-step L2 traffic: 101*32MB ~= 3.2GB)
// and pre-swizzle it into v_wmma_f32_16x16x32_f16 A-fragment order so the hot
// loop issues coalesced global_load_b128 + ds_load_b128 + v_wmma only.
// The h vector is broadcast across all 16 WMMA N-columns (compute waste is
// free when memory bound; fp32 accumulation preserved).
//
// The GEMV loop is explicitly software-pipelined 2 deep (even/odd fragment
// buffers) so each WMMA consumes loads issued a full stage earlier -> partial
// loadcnt waits instead of s_wait_loadcnt 0, critical at ~1 wave/SIMD
// occupancy. global_prefetch_b8 hints pull the L2-resident stream ahead.
// ---------------------------------------------------------------------------

typedef __attribute__((ext_vector_type(16))) _Float16 v16h;
typedef __attribute__((ext_vector_type(8)))  float    v8f;

#define HID   2048
#define GATES 8192          // 4*HID
#define INSZ  1024
#define OUTSZ 1024
#define NSTEP 101           // M_STEPS + 1
#define KTILES 64           // 2048 / 32
#define ROWTILES 512        // 8192 / 16

// ---------------------------------------------------------------------------
// Kernel 1: convert W_hh fp32 -> f16, pre-swizzled into WMMA A-fragment order.
// Dest layout: tile = (tileRow[0..511], kChunk[0..63]); within a tile, 32
// lanes x 16 halves, lane-major contiguous (so the GEMV loads 32B/lane
// contiguous & coalesced). Element map (16-bit A, 16x32; contiguous k, same order as B):
//   row = tileRow*16 + (lane & 15)
//   k   = kChunk*32 + (lane>=16 ? 16 : 0) + j
// ---------------------------------------------------------------------------
__global__ void __launch_bounds__(256)
convert_whh_kernel(const float* __restrict__ Whh, _Float16* __restrict__ Wf16) {
  int g    = blockIdx.x * 256 + threadIdx.x;   // one thread per lane-fragment
  int lane = g & 31;
  int tile = g >> 5;                           // 0..32767
  int kc   = tile & (KTILES - 1);
  int tr   = tile >> 6;                        // 0..511
  int row   = tr * 16 + (lane & 15);
  int kbase = kc * 32 + ((lane >> 4) << 4);
  const float* src = Whh + (size_t)row * HID + kbase;
  _Float16*    dst = Wf16 + (size_t)tile * 512 + lane * 16;
#pragma unroll
  for (int j = 0; j < 16; ++j) {
    int k = j;
    dst[j] = (_Float16)src[k];
  }
}

// ---------------------------------------------------------------------------
// Kernel 2: ih_x = W_ih[:,1:] @ x + b_ih + b_hh  (one-time, 8192 rows).
// Also extracts w_flag = W_ih[:,0] into a contiguous buffer.
// ---------------------------------------------------------------------------
__global__ void __launch_bounds__(256)
ihx_kernel(const float* __restrict__ W_ih, const float* __restrict__ x,
           const float* __restrict__ b_ih, const float* __restrict__ b_hh,
           float* __restrict__ ihx, float* __restrict__ wflag) {
  int row = blockIdx.x;
  const float* wr = W_ih + (size_t)row * (INSZ + 1);
  float s = 0.f;
  for (int k = threadIdx.x; k < INSZ; k += 256) s += wr[1 + k] * x[k];
  __shared__ float red[256];
  red[threadIdx.x] = s; __syncthreads();
  for (int o = 128; o > 0; o >>= 1) {
    if (threadIdx.x < o) red[threadIdx.x] += red[threadIdx.x + o];
    __syncthreads();
  }
  if (threadIdx.x == 0) {
    ihx[row]   = red[0] + b_ih[row] + b_hh[row];
    wflag[row] = wr[0];
  }
}

// ---------------------------------------------------------------------------
// Kernel 3: one LSTM step. 64 blocks x 128 threads (4 waves).
// Wave w computes gate region w (i/f/g/o) for 32 rows via WMMA, then 32
// threads fuse the cell update. B = h-chunk broadcast over all 16 columns
// (contiguous k order, same as A: lanes 0-15 hold K 0-15, lanes 16-31 K 16-31).
// ---------------------------------------------------------------------------
__global__ void __launch_bounds__(128)
step_kernel(const _Float16* __restrict__ Wf16,
            const float* __restrict__ ihx, const float* __restrict__ wflag,
            float flag,
            const float* __restrict__ h_in, const float* __restrict__ c_in,
            float* __restrict__ h_out, float* __restrict__ c_out) {
  __shared__ alignas(32) _Float16 hsh[HID];
  __shared__ float gbuf[4][32];

  const int tid  = threadIdx.x;
  const int lane = tid & 31;
  const int wv   = tid >> 5;                    // gate region 0..3

  // Stage h (fp32 -> f16) into LDS; all 128 threads, no divergence.
  for (int i = tid; i < HID; i += 128) hsh[i] = (_Float16)h_in[i];
  __syncthreads();

  // Two 16-row tiles per wave: tile rows within full 8192-row gate matrix.
  const int tr0 = wv * 128 + blockIdx.x * 2;    // region wv = tile rows [wv*128, wv*128+128)
  const _Float16* p0 = Wf16 + (size_t)tr0 * KTILES * 512 + lane * 16;
  const _Float16* p1 = p0 + (size_t)KTILES * 512;
  const int boff = (lane >> 4) << 4;            // lanes 0-15 -> k 0-15, 16-31 -> k 16-31

  v8f acc0 = {}; v8f acc1 = {};

  // --- software-pipelined GEMV: even/odd stages, loads one stage ahead ----
  v16h a0e = *(const v16h*)(p0);
  v16h a1e = *(const v16h*)(p1);
  v16h be  = *(const v16h*)(&hsh[boff]);

  for (int kc = 0; kc < KTILES; kc += 2) {
    // issue odd-stage loads before consuming even-stage fragments
    v16h a0o = *(const v16h*)(p0 + (size_t)(kc + 1) * 512);
    v16h a1o = *(const v16h*)(p1 + (size_t)(kc + 1) * 512);
    v16h bo  = *(const v16h*)(&hsh[(kc + 1) * 32 + boff]);

    // long-range hint: pull the L2-resident stream toward the WGP
    // (global_prefetch_b8; speculative, over-run past tile end is harmless)
    __builtin_prefetch(p0 + (size_t)(kc + 8) * 512, 0, 3);
    __builtin_prefetch(p1 + (size_t)(kc + 8) * 512, 0, 3);

    acc0 = __builtin_amdgcn_wmma_f32_16x16x32_f16(false, a0e, false, be,
                                                  (short)0, acc0, false, false);
    acc1 = __builtin_amdgcn_wmma_f32_16x16x32_f16(false, a1e, false, be,
                                                  (short)0, acc1, false, false);

    // issue next even-stage loads before consuming odd-stage fragments
    if (kc + 2 < KTILES) {
      a0e = *(const v16h*)(p0 + (size_t)(kc + 2) * 512);
      a1e = *(const v16h*)(p1 + (size_t)(kc + 2) * 512);
      be  = *(const v16h*)(&hsh[(kc + 2) * 32 + boff]);
    }

    acc0 = __builtin_amdgcn_wmma_f32_16x16x32_f16(false, a0o, false, bo,
                                                  (short)0, acc0, false, false);
    acc1 = __builtin_amdgcn_wmma_f32_16x16x32_f16(false, a1o, false, bo,
                                                  (short)0, acc1, false, false);
  }

  // D column 0 lives in lanes 0 (M=0..7) and 16 (M=8..15).
  if ((lane & 15) == 0) {
    int mh = (lane >> 4) * 8;
#pragma unroll
    for (int r = 0; r < 8; ++r) {
      gbuf[wv][mh + r]      = acc0[r];
      gbuf[wv][16 + mh + r] = acc1[r];
    }
  }
  __syncthreads();

  // Fused cell update for this block's 32 hidden rows.
  if (tid < 32) {
    int row = blockIdx.x * 32 + tid;
    float g0 = ihx[row           ] + gbuf[0][tid];
    float g1 = ihx[HID     + row ] + gbuf[1][tid];
    float g2 = ihx[2*HID   + row ] + gbuf[2][tid];
    float g3 = ihx[3*HID   + row ] + gbuf[3][tid];
    if (flag != 0.f) {                          // uniform branch (t==0 only)
      g0 += flag * wflag[row];
      g1 += flag * wflag[HID   + row];
      g2 += flag * wflag[2*HID + row];
      g3 += flag * wflag[3*HID + row];
    }
    float ig = 1.f / (1.f + __expf(-g0));
    float fg = 1.f / (1.f + __expf(-g1));
    float gg = tanhf(g2);
    float og = 1.f / (1.f + __expf(-g3));
    float c  = fg * c_in[row] + ig * gg;
    float h  = og * tanhf(c);
    h_out[row] = h;
    c_out[row] = c;
  }
}

// ---------------------------------------------------------------------------
// Kernel 4: p[t] = sigmoid(H[t] . w_halt + b_halt), one block per t.
// ---------------------------------------------------------------------------
__global__ void __launch_bounds__(256)
halt_kernel(const float* __restrict__ H, const float* __restrict__ w_halt,
            const float* __restrict__ b_halt, float* __restrict__ p) {
  int t = blockIdx.x;
  const float* h = H + (size_t)t * HID;
  float s = 0.f;
  for (int k = threadIdx.x; k < HID; k += 256) s += h[k] * w_halt[k];
  __shared__ float red[256];
  red[threadIdx.x] = s; __syncthreads();
  for (int o = 128; o > 0; o >>= 1) {
    if (threadIdx.x < o) red[threadIdx.x] += red[threadIdx.x + o];
    __syncthreads();
  }
  if (threadIdx.x == 0) p[t] = 1.f / (1.f + __expf(-(red[0] + b_halt[0])));
}

// ---------------------------------------------------------------------------
// Kernel 5: serial cumsum (101 values), halting index n, weights w[t], ponder.
// ---------------------------------------------------------------------------
__global__ void weights_kernel(const float* __restrict__ p,
                               float* __restrict__ w,
                               float* __restrict__ ponder_out) {
  if (threadIdx.x == 0) {
    float cs[NSTEP];
    float csum = 0.f;
    for (int t = 0; t < NSTEP; ++t) { csum += p[t]; cs[t] = csum; }
    int n = 100;                                  // M_STEPS default (not halted)
    for (int t = 0; t < NSTEP; ++t) {
      if (cs[t] >= 1.f - 0.01f) { n = t; break; }
    }
    float r = (n > 0) ? (1.f - (cs[n] - p[n])) : 1.f;
    for (int t = 0; t < NSTEP; ++t)
      w[t] = (t < n) ? p[t] : ((t == n) ? r : 0.f);
    ponder_out[0] = (float)n + 1.f + r;
  }
}

// ---------------------------------------------------------------------------
// Kernel 6: h_out[j] = sum_t w[t]*H[t][j]; c_out likewise. Writes d_out[1024..5119]
// and a ws copy for the final GEMV.
// ---------------------------------------------------------------------------
__global__ void __launch_bounds__(256)
hcout_kernel(const float* __restrict__ H, const float* __restrict__ C,
             const float* __restrict__ w, float* __restrict__ hc,
             float* __restrict__ dout) {
  int j = blockIdx.x * 256 + threadIdx.x;       // 0..4095 (h then c)
  const float* S = (j < HID) ? H : C;
  int col = j & (HID - 1);
  float s = 0.f;
  for (int t = 0; t < NSTEP; ++t) s += w[t] * S[(size_t)t * HID + col];
  hc[j] = s;
  dout[OUTSZ + j] = s;
}

// ---------------------------------------------------------------------------
// Kernel 7: output = W_out @ h_out + b_out  (halting weights sum to 1
// analytically, so the per-step output GEMVs collapse into one GEMV).
// ---------------------------------------------------------------------------
__global__ void __launch_bounds__(256)
out_kernel(const float* __restrict__ W_out, const float* __restrict__ b_out,
           const float* __restrict__ hvec, float* __restrict__ dout) {
  int row = blockIdx.x;
  const float* wr = W_out + (size_t)row * HID;
  float s = 0.f;
  for (int k = threadIdx.x; k < HID; k += 256) s += wr[k] * hvec[k];
  __shared__ float red[256];
  red[threadIdx.x] = s; __syncthreads();
  for (int o = 128; o > 0; o >>= 1) {
    if (threadIdx.x < o) red[threadIdx.x] += red[threadIdx.x + o];
    __syncthreads();
  }
  if (threadIdx.x == 0) dout[row] = red[0] + b_out[row];
}

// ---------------------------------------------------------------------------
// Host launcher. Workspace layout (bytes), total ~33.7 MiB:
//   [0)              Wf16        8192*2048*2 = 33,554,432
//   [33,554,432)     ihx         8192*4
//   [33,587,200)     wflag       8192*4
//   [33,619,968)     H           101*2048*4  = 827,392
//   [34,447,360)     C           101*2048*4  = 827,392
//   [35,274,752)     p           101*4  (padded to 512)
//   [35,275,264)     w           101*4  (padded to 512)
//   [35,275,776)     hc          4096*4
// ---------------------------------------------------------------------------
extern "C" void kernel_launch(void* const* d_in, const int* in_sizes, int n_in,
                              void* d_out, int out_size, void* d_ws, size_t ws_size,
                              hipStream_t stream) {
  const float* x      = (const float*)d_in[0];
  const float* h0     = (const float*)d_in[1];
  const float* c0     = (const float*)d_in[2];
  const float* W_ih   = (const float*)d_in[3];
  const float* b_ih   = (const float*)d_in[4];
  const float* W_hh   = (const float*)d_in[5];
  const float* b_hh   = (const float*)d_in[6];
  const float* w_halt = (const float*)d_in[7];
  const float* b_halt = (const float*)d_in[8];
  const float* W_out  = (const float*)d_in[9];
  const float* b_out  = (const float*)d_in[10];
  float* out = (float*)d_out;

  char* ws = (char*)d_ws;
  _Float16* Wf16 = (_Float16*)ws;
  float* ihx   = (float*)(ws + 33554432u);
  float* wflag = (float*)(ws + 33587200u);
  float* H     = (float*)(ws + 33619968u);
  float* C     = (float*)(ws + 34447360u);
  float* p     = (float*)(ws + 35274752u);
  float* wt    = (float*)(ws + 35275264u);
  float* hc    = (float*)(ws + 35275776u);

  // One-time weight prep (f16 swizzle stays hot in L2 across all 101 steps).
  convert_whh_kernel<<<4096, 256, 0, stream>>>(W_hh, Wf16);
  ihx_kernel<<<GATES, 256, 0, stream>>>(W_ih, x, b_ih, b_hh, ihx, wflag);

  // 101 dependent recurrent steps (stream-ordered; graph replay amortizes launch).
  for (int t = 0; t < NSTEP; ++t) {
    const float* hin = (t == 0) ? h0 : H + (size_t)(t - 1) * HID;
    const float* cin = (t == 0) ? c0 : C + (size_t)(t - 1) * HID;
    step_kernel<<<64, 128, 0, stream>>>(Wf16, ihx, wflag,
                                        (t == 0) ? 1.f : 0.f,
                                        hin, cin,
                                        H + (size_t)t * HID,
                                        C + (size_t)t * HID);
  }

  // Halting epilogue.
  halt_kernel<<<NSTEP, 256, 0, stream>>>(H, w_halt, b_halt, p);
  weights_kernel<<<1, 32, 0, stream>>>(p, wt, out + 5120);
  hcout_kernel<<<16, 256, 0, stream>>>(H, C, wt, hc, out);
  out_kernel<<<OUTSZ, 256, 0, stream>>>(W_out, b_out, hc, out);
}
